// Attention_70231305224843
// MI455X (gfx1250) — matrix operand order repaired and checked by reading, not compiled, not run on hardware
//
#include <hip/hip_runtime.h>

typedef unsigned short u16;
typedef unsigned int   u32;
typedef __attribute__((ext_vector_type(16))) __bf16 v16bf;
typedef __attribute__((ext_vector_type(8)))  float  v8f;
typedef __attribute__((ext_vector_type(8)))  u16    u16x8;
typedef __attribute__((ext_vector_type(4)))  u32    u32x4;
typedef __attribute__((ext_vector_type(8)))  int    i32x8;
typedef __attribute__((ext_vector_type(4)))  int    i32x4;

#define TOK     4096      // B*N
#define DMODEL  1024
#define NSEQ    2048
#define NH      16
#define DHD     64
#define NEGBIG  -1000000000.0f

#define KROW    72        // LDS row stride (u16) for TDM-padded K tile: 128B+16B

union V16U { v16bf v; u16x8 h[2]; };

__device__ __forceinline__ v16bf ldA(const u16* p0, const u16* p1) {
  V16U t;
  t.h[0] = *reinterpret_cast<const u16x8*>(p0);
  t.h[1] = *reinterpret_cast<const u16x8*>(p1);
  return t.v;
}
__device__ __forceinline__ v16bf ld16(const u16* p) { return ldA(p, p + 8); }

__device__ __forceinline__ u16 f2bf(float f) {
  u32 u = __float_as_uint(f);
  u = u + 0x7FFFu + ((u >> 16) & 1u);   // round-to-nearest-even
  return (u16)(u >> 16);
}

__device__ __forceinline__ v8f wmma_bf16(v16bf a, v16bf b, v8f c) {
  return __builtin_amdgcn_wmma_f32_16x16x32_bf16(false, a, false, b, (short)0, c,
                                                 false, false);
}

__device__ __forceinline__ float redmax16(float v) {
  v = fmaxf(v, __shfl_xor(v, 1, 32));
  v = fmaxf(v, __shfl_xor(v, 2, 32));
  v = fmaxf(v, __shfl_xor(v, 4, 32));
  v = fmaxf(v, __shfl_xor(v, 8, 32));
  return v;
}
__device__ __forceinline__ float redsum16(float v) {
  v += __shfl_xor(v, 1, 32);
  v += __shfl_xor(v, 2, 32);
  v += __shfl_xor(v, 4, 32);
  v += __shfl_xor(v, 8, 32);
  return v;
}

// TDM: load a 32x64 bf16 tile (rows of 128B) into LDS with 16B row padding
// (LDS row stride 144B).  2D descriptor, data_size=2B, tensor_dim0_stride=64.
__device__ __forceinline__ void tdm_load_k32x64(u32 lds_off, const u16* gptr) {
  const unsigned long long ga = (unsigned long long)(size_t)gptr;
  u32x4 g0;
  g0[0] = 1u;                                       // count=1 (valid user D#)
  g0[1] = lds_off;                                  // lds_addr
  g0[2] = (u32)(ga & 0xFFFFFFFFu);                  // global_addr[31:0]
  g0[3] = (u32)((ga >> 32) & 0x01FFFFFFu) | 0x80000000u; // addr[56:32]|type=2
  i32x8 g1;
  g1[0] = (1 << 16)      // data_size = 1 -> 2 bytes
        | (1 << 20)      // pad_enable
        | (4 << 22)      // pad_interval code 4 -> every 32 DWORDs (128B)
        | (3 << 25);     // pad_amount  code 3 -> 4 DWORDs (16B)
  g1[1] = (64 << 16);    // tensor_dim0 = 64 (lo16 at bits 63:48)
  g1[2] = (int)(2048u << 16);  // tensor_dim0 hi=0 | tensor_dim1 lo16 = 2048
  g1[3] = (64 << 16);    // tensor_dim1 hi=0 | tile_dim0 = 64
  g1[4] = 32;            // tile_dim1 = 32, tile_dim2 = 0
  g1[5] = 64;            // tensor_dim0_stride lo32 = 64
  g1[6] = 0;             // stride hi16 | tensor_dim1_stride lo16
  g1[7] = 0;
  i32x4 z4 = {0, 0, 0, 0};
#if __clang_major__ >= 23
  i32x8 z8 = {0, 0, 0, 0, 0, 0, 0, 0};
  __builtin_amdgcn_tensor_load_to_lds(g0, g1, z4, z4, z8, 0);
#else
  __builtin_amdgcn_tensor_load_to_lds(g0, g1, z4, z4, 0);
#endif
}

// ---------------------------------------------------------------- f32 -> bf16
__global__ void cvt_bf16_kernel(const float* __restrict__ src,
                                u16* __restrict__ dst, int n) {
  int i = blockIdx.x * blockDim.x + threadIdx.x;
  int stride = gridDim.x * blockDim.x;
  for (; i < n; i += stride) dst[i] = f2bf(src[i]);
}

// ---------------------------------- f32 [k][n] -> bf16 transposed [n][k] tile
__global__ __launch_bounds__(256) void cvt_t_bf16_kernel(
    const float* __restrict__ src, u16* __restrict__ dst) {
  __shared__ float tile[32][33];
  const int tx = threadIdx.x & 31;
  const int tg = threadIdx.x >> 5;       // 8 groups of 4 rows
  const int r0 = blockIdx.y * 32;        // k
  const int c0 = blockIdx.x * 32;        // n
#pragma unroll
  for (int i = 0; i < 4; ++i) {
    const int r = tg * 4 + i;
    tile[r][tx] = src[(size_t)(r0 + r) * DMODEL + c0 + tx];
  }
  __syncthreads();
#pragma unroll
  for (int i = 0; i < 4; ++i) {
    const int c = tg * 4 + i;
    dst[(size_t)(c0 + c) * DMODEL + r0 + tx] = f2bf(tile[tx][c]);
  }
}

// ------------------------- QKV GEMM (direct-global B) + bias + RoPE + pack
// grid (16, 16, 3), block 256.  Wave tile: 32 rows x 64 cols (2 A per B).
__global__ __launch_bounds__(256) void qkv_rope_kernel(
    const u16* __restrict__ xb,
    const u16* __restrict__ wqT, const u16* __restrict__ wkT,
    const u16* __restrict__ wvT,
    const float* __restrict__ bq, const float* __restrict__ bk,
    const float* __restrict__ bv,
    const float* __restrict__ ropec, const float* __restrict__ ropes,
    u16* __restrict__ qo, u16* __restrict__ ko, u16* __restrict__ voT) {
  const int lane = threadIdx.x & 31;
  const int wave = threadIdx.x >> 5;
  const int h16  = lane >> 4;
  const int l16  = lane & 15;
  const int which = blockIdx.z;
  const u16*  WT   = (which == 0) ? wqT : ((which == 1) ? wkT : wvT);
  const float* bias = (which == 0) ? bq : ((which == 1) ? bk : bv);
  const int m0 = blockIdx.x * 256 + wave * 32;
  const int n0 = blockIdx.y * 64;

  v8f acc[2][4];
#pragma unroll
  for (int mi = 0; mi < 2; ++mi)
#pragma unroll
    for (int t = 0; t < 4; ++t)
      acc[mi][t] = (v8f){0.f,0.f,0.f,0.f,0.f,0.f,0.f,0.f};

  const u16* xrow0 = xb + (size_t)(m0 + l16) * DMODEL;
  const u16* xrow1 = xb + (size_t)(m0 + 16 + l16) * DMODEL;
  const u16* brow[4];
#pragma unroll
  for (int t = 0; t < 4; ++t)
    brow[t] = WT + (size_t)(n0 + t * 16 + l16) * DMODEL;

  for (int k0 = 0; k0 < DMODEL; k0 += 32) {
    __builtin_prefetch(xrow0 + k0 + 128, 0, 3);
    __builtin_prefetch(xrow1 + k0 + 128, 0, 3);
    __builtin_prefetch(brow[0] + k0 + 128, 0, 3);
    const v16bf a0 = ldA(xrow0 + k0 + h16 * 8, xrow0 + k0 + 16 + h16 * 8);
    const v16bf a1 = ldA(xrow1 + k0 + h16 * 8, xrow1 + k0 + 16 + h16 * 8);
#pragma unroll
    for (int t = 0; t < 4; ++t) {
      const v16bf bt = ldA(brow[t] + k0 + h16 * 8, brow[t] + k0 + 16 + h16 * 8);
      acc[0][t] = wmma_bf16(a0, bt, acc[0][t]);
      acc[1][t] = wmma_bf16(a1, bt, acc[1][t]);
    }
  }

#pragma unroll
  for (int mi = 0; mi < 2; ++mi) {
#pragma unroll
    for (int t = 0; t < 4; ++t) {
      const int col = n0 + t * 16 + l16;
      const float bcol = bias[col];
#pragma unroll
      for (int r = 0; r < 8; ++r) {
        const int row = m0 + mi * 16 + r + 8 * h16;  // token in [0, 4096)
        float val = acc[mi][t][r] + bcol;
        if (which < 2) {                             // RoPE on q and k
          const float cs = ropec[(size_t)row * DMODEL + col];
          const float sn = ropes[(size_t)row * DMODEL + col];
          const float partner = __shfl_xor(val, 1, 32);
          val = (col & 1) ? (val * cs + partner * sn)
                          : (val * cs - partner * sn);
        }
        const int b  = row >> 11;
        const int n  = row & (NSEQ - 1);
        const int hh = col >> 6;
        const int d  = col & 63;
        if (which == 2)   // V stored transposed: [B,H,DH,N]
          voT[(((size_t)(b * NH + hh) * DHD) + d) * NSEQ + n] = f2bf(val);
        else if (which == 0)
          qo[(((size_t)(b * NH + hh) * NSEQ) + n) * DHD + d] = f2bf(val);
        else
          ko[(((size_t)(b * NH + hh) * NSEQ) + n) * DHD + d] = f2bf(val);
      }
    }
  }
}

// ------------------------------------------------------------ flash attention
// grid 512 = B*H*(N/128), block 256 (8 waves x 16 q-rows).
// K chunk staged into LDS by the Tensor Data Mover (double-buffered);
// V read directly from its transposed global layout.
__global__ __launch_bounds__(256) void flash_attn_kernel(
    const u16* __restrict__ qb, const u16* __restrict__ kb,
    const u16* __restrict__ vbT, const int* __restrict__ lens,
    u16* __restrict__ ctxb) {
  const int lane = threadIdx.x & 31;
  const int wave = threadIdx.x >> 5;
  const int h16  = lane >> 4;
  const int l16  = lane & 15;

  const int blk  = blockIdx.x;
  const int qblk = blk & 15;
  const int head = (blk >> 4) & 15;
  const int b    = blk >> 8;
  const int len  = lens[b];

  const u16* qbase = qb  + (size_t)(b * NH + head) * NSEQ * DHD;
  const u16* kbase = kb  + (size_t)(b * NH + head) * NSEQ * DHD;
  const u16* vbase = vbT + (size_t)(b * NH + head) * DHD * NSEQ;

  const int q0 = qblk * 128 + wave * 16;

  // Q tile as two A operands (d = 0..31 and 32..63), resident in VGPRs
  const u16* qrow = qbase + (size_t)(q0 + l16) * DHD;
  const v16bf qa0 = ldA(qrow +      h16 * 8, qrow + 16 + h16 * 8);
  const v16bf qa1 = ldA(qrow + 32 + h16 * 8, qrow + 48 + h16 * 8);

  __shared__ __align__(16) u16 klds[2][32 * KROW];  // TDM-padded K tiles
  __shared__ __align__(16) u16 pl[8 * 16 * 40];     // per-wave P tiles [m][k]
  u16* mypl = &pl[wave * 16 * 40];

  v8f o[4];
#pragma unroll
  for (int t = 0; t < 4; ++t) o[t] = (v8f){0.f,0.f,0.f,0.f,0.f,0.f,0.f,0.f};
  float mrun[8], lrun[8];
#pragma unroll
  for (int r = 0; r < 8; ++r) { mrun[r] = -3.0e38f; lrun[r] = 0.0f; }

  // V row pointers (lane = output column d of each 16-wide tile)
  const u16* vrow[4];
#pragma unroll
  for (int t = 0; t < 4; ++t)
    vrow[t] = vbase + (size_t)(t * 16 + l16) * NSEQ;

  // prologue: TDM chunk 0 into buffer 0
  if (wave == 0) {
    tdm_load_k32x64((u32)(size_t)(void*)&klds[0][0], kbase);
    __builtin_amdgcn_s_wait_tensorcnt(0);
  }
  __syncthreads();

  int cur = 0;
  for (int kc = 0; kc < NSEQ; kc += 32) {
    if (wave == 0 && (kc + 32) < NSEQ)
      tdm_load_k32x64((u32)(size_t)(void*)&klds[cur ^ 1][0],
                      kbase + (size_t)(kc + 32) * DHD);

    __builtin_prefetch(vrow[0] + kc + 32, 0, 3);
    __builtin_prefetch(vrow[2] + kc + 32, 0, 3);

    // scores for two 16-key tiles; K tile read from LDS (TDM-staged)
    const u16* kr0 = &klds[cur][(l16)      * KROW];
    const u16* kr1 = &klds[cur][(16 + l16) * KROW];
    v8f s0 = (v8f){0.f,0.f,0.f,0.f,0.f,0.f,0.f,0.f};
    v8f s1 = (v8f){0.f,0.f,0.f,0.f,0.f,0.f,0.f,0.f};
    s0 = wmma_bf16(qa0, ldA(kr0 +      h16 * 8, kr0 + 16 + h16 * 8), s0);
    s0 = wmma_bf16(qa1, ldA(kr0 + 32 + h16 * 8, kr0 + 48 + h16 * 8), s0);
    s1 = wmma_bf16(qa0, ldA(kr1 +      h16 * 8, kr1 + 16 + h16 * 8), s1);
    s1 = wmma_bf16(qa1, ldA(kr1 + 32 + h16 * 8, kr1 + 48 + h16 * 8), s1);

    const float msk0 = (kc + l16      < len) ? 0.0f : NEGBIG;
    const float msk1 = (kc + 16 + l16 < len) ? 0.0f : NEGBIG;

#pragma unroll
    for (int r = 0; r < 8; ++r) {
      const float a0 = s0[r] + msk0;
      const float a1 = s1[r] + msk1;
      const float cmax  = redmax16(fmaxf(a0, a1));
      const float mnew  = fmaxf(mrun[r], cmax);
      const float alpha = __expf(mrun[r] - mnew);
      const float p0 = __expf(a0 - mnew);
      const float p1 = __expf(a1 - mnew);
      lrun[r] = lrun[r] * alpha + redsum16(p0 + p1);
      mrun[r] = mnew;
#pragma unroll
      for (int t = 0; t < 4; ++t) o[t][r] *= alpha;
      const int prow = (r + 8 * h16) * 40;
      mypl[prow + l16]      = f2bf(p0);
      mypl[prow + 16 + l16] = f2bf(p1);
    }

    // intra-wave P transpose: DS ops of one wave are in-order; wait + fence
    asm volatile("s_wait_dscnt 0x0" ::: "memory");
    const v16bf pa = ldA(mypl + l16 * 40 + h16 * 8,
                         mypl + l16 * 40 + 16 + h16 * 8);
#pragma unroll
    for (int t = 0; t < 4; ++t)
      o[t] = wmma_bf16(pa, ldA(vrow[t] + kc + h16 * 8, vrow[t] + kc + 16 + h16 * 8), o[t]);

    if (wave == 0) __builtin_amdgcn_s_wait_tensorcnt(0);
    __syncthreads();
    cur ^= 1;
  }

#pragma unroll
  for (int r = 0; r < 8; ++r) {
    const float inv = 1.0f / lrun[r];
    const int q = q0 + r + 8 * h16;
    const size_t base = ((size_t)b * NSEQ + q) * (NH * DHD) + head * DHD;
#pragma unroll
    for (int t = 0; t < 4; ++t)
      ctxb[base + t * 16 + l16] = f2bf(o[t][r] * inv);
  }
}

// ------------------------- out projection (direct-global B) + bias + row mask
// grid (16, 16), block 256.  Wave tile: 32 rows x 64 cols.
__global__ __launch_bounds__(256) void oproj_kernel(
    const u16* __restrict__ ctxb, const u16* __restrict__ woT,
    const float* __restrict__ bo, const int* __restrict__ lens,
    float* __restrict__ out) {
  const int lane = threadIdx.x & 31;
  const int wave = threadIdx.x >> 5;
  const int h16  = lane >> 4;
  const int l16  = lane & 15;
  const int m0 = blockIdx.x * 256 + wave * 32;
  const int n0 = blockIdx.y * 64;
  const int len0 = lens[0], len1 = lens[1];

  v8f acc[2][4];
#pragma unroll
  for (int mi = 0; mi < 2; ++mi)
#pragma unroll
    for (int t = 0; t < 4; ++t)
      acc[mi][t] = (v8f){0.f,0.f,0.f,0.f,0.f,0.f,0.f,0.f};

  const u16* arow0 = ctxb + (size_t)(m0 + l16) * DMODEL;
  const u16* arow1 = ctxb + (size_t)(m0 + 16 + l16) * DMODEL;
  const u16* brow[4];
#pragma unroll
  for (int t = 0; t < 4; ++t)
    brow[t] = woT + (size_t)(n0 + t * 16 + l16) * DMODEL;

  for (int k0 = 0; k0 < DMODEL; k0 += 32) {
    __builtin_prefetch(arow0 + k0 + 128, 0, 3);
    __builtin_prefetch(arow1 + k0 + 128, 0, 3);
    const v16bf a0 = ldA(arow0 + k0 + h16 * 8, arow0 + k0 + 16 + h16 * 8);
    const v16bf a1 = ldA(arow1 + k0 + h16 * 8, arow1 + k0 + 16 + h16 * 8);
#pragma unroll
    for (int t = 0; t < 4; ++t) {
      const v16bf bt = ldA(brow[t] + k0 + h16 * 8, brow[t] + k0 + 16 + h16 * 8);
      acc[0][t] = wmma_bf16(a0, bt, acc[0][t]);
      acc[1][t] = wmma_bf16(a1, bt, acc[1][t]);
    }
  }

#pragma unroll
  for (int mi = 0; mi < 2; ++mi) {
#pragma unroll
    for (int t = 0; t < 4; ++t) {
      const int col = n0 + t * 16 + l16;
      const float bcol = bo[col];
#pragma unroll
      for (int r = 0; r < 8; ++r) {
        const int row = m0 + mi * 16 + r + 8 * h16;
        const int bb = row >> 11;
        const int n  = row & (NSEQ - 1);
        const int len = bb ? len1 : len0;
        const float val = acc[mi][t][r] + bcol;
        out[(size_t)row * DMODEL + col] = (n < len) ? val : 0.0f;
      }
    }
  }
}

// ----------------------------------------------------------------- launcher
extern "C" void kernel_launch(void* const* d_in, const int* in_sizes, int n_in,
                              void* d_out, int out_size, void* d_ws,
                              size_t ws_size, hipStream_t stream) {
  const float* x     = (const float*)d_in[0];
  const float* ropec = (const float*)d_in[1];
  const float* ropes = (const float*)d_in[2];
  const int*   lens  = (const int*)d_in[3];
  const float* Wq    = (const float*)d_in[4];
  const float* bq    = (const float*)d_in[5];
  const float* Wk    = (const float*)d_in[6];
  const float* bk    = (const float*)d_in[7];
  const float* Wv    = (const float*)d_in[8];
  const float* bv    = (const float*)d_in[9];
  const float* Wo    = (const float*)d_in[10];
  const float* bo    = (const float*)d_in[11];
  float* out = (float*)d_out;

  u16* ws  = (u16*)d_ws;
  u16* xb  = ws;                                  // 4096*1024
  u16* wqT = xb  + (size_t)TOK * DMODEL;          // 1024*1024 each, transposed
  u16* wkT = wqT + (size_t)DMODEL * DMODEL;
  u16* wvT = wkT + (size_t)DMODEL * DMODEL;
  u16* woT = wvT + (size_t)DMODEL * DMODEL;
  u16* qbx = woT + (size_t)DMODEL * DMODEL;       // [B,H,N,DH]
  u16* kbx = qbx + (size_t)TOK * DMODEL;          // [B,H,N,DH]
  u16* vtx = kbx + (size_t)TOK * DMODEL;          // [B,H,DH,N] (transposed)
  u16* ctx = vtx + (size_t)TOK * DMODEL;          // [B*N, H*DH]

  cvt_bf16_kernel<<<2048, 256, 0, stream>>>(x, xb, TOK * DMODEL);
  cvt_t_bf16_kernel<<<dim3(32, 32), 256, 0, stream>>>(Wq, wqT);
  cvt_t_bf16_kernel<<<dim3(32, 32), 256, 0, stream>>>(Wk, wkT);
  cvt_t_bf16_kernel<<<dim3(32, 32), 256, 0, stream>>>(Wv, wvT);
  cvt_t_bf16_kernel<<<dim3(32, 32), 256, 0, stream>>>(Wo, woT);

  qkv_rope_kernel<<<dim3(16, 16, 3), 256, 0, stream>>>(
      xb, wqT, wkT, wvT, bq, bk, bv, ropec, ropes, qbx, kbx, vtx);

  flash_attn_kernel<<<512, 256, 0, stream>>>(qbx, kbx, vtx, lens, ctx);

  oproj_kernel<<<dim3(16, 16), 256, 0, stream>>>(ctx, woT, bo, lens, out);
}
